// CosformerAttention_55362128445664
// MI455X (gfx1250) — matrix operand order repaired and checked by reading, not compiled, not run on hardware
//
#include <hip/hip_runtime.h>
#include <hip/hip_bf16.h>
#include <math.h>

typedef __attribute__((ext_vector_type(16))) __bf16 v16bf;
typedef __attribute__((ext_vector_type(8)))  __bf16 v8bf;
typedef __attribute__((ext_vector_type(4)))  __bf16 v4bf;
typedef __attribute__((ext_vector_type(8)))  float  v8f;
typedef __attribute__((ext_vector_type(4)))  float  v4f;
typedef __attribute__((ext_vector_type(4)))  int    i32x4;

#define AS1 __attribute__((address_space(1)))
#define AS3 __attribute__((address_space(3)))

#define L_SEQ 4096
#define BATCH 4
#define EMB   1024
#define HD    64
#define NHEAD 64              /* B*H */
#define ROWS  16384           /* L*B */
#define EPS_Z 1e-4f

// ---- CDNA5 async global->LDS path (guarded: falls back to sync copy) ----
#if __has_builtin(__builtin_amdgcn_global_load_async_to_lds_b128) && \
    __has_builtin(__builtin_amdgcn_s_wait_asynccnt)
#define ASYNC_LDS 1
#else
#define ASYNC_LDS 0
#endif

__device__ __forceinline__ void cp16_g2l(const __bf16* g, __bf16* l) {
#if ASYNC_LDS
    // signature (from compiler diagnostic): (v4i AS1*, v4i AS3*, imm, imm)
    __builtin_amdgcn_global_load_async_to_lds_b128(
        (AS1 i32x4*)g, (AS3 i32x4*)l, 0, 0);
#else
    *(v8bf*)l = *(const v8bf*)g;
#endif
}

__device__ __forceinline__ void cp_fence() {
#if ASYNC_LDS
    __builtin_amdgcn_s_wait_asynccnt(0);
#endif
}

__device__ __forceinline__ void sincos_pos(int l, float* s, float* c) {
    float ang = 1.57079632679489662f * (float)(l + 1) * (1.0f / (float)L_SEQ);
    *s = __sinf(ang);
    *c = __cosf(ang);
}

// ---------------------------------------------------------------------------
// Kernel 0: fp32 -> bf16 convert (one-time; makes GEMM staging a pure
// byte-copy so the async-to-LDS path applies and hot-loop VALU cvt vanishes).
// ---------------------------------------------------------------------------
__global__ __launch_bounds__(256)
void cvt_kernel(const float* __restrict__ in, __bf16* __restrict__ out, int n4) {
    int i = blockIdx.x * 256 + threadIdx.x;
    if (i < n4) {
        v4f x = *(const v4f*)(in + (size_t)i * 4);
        v4bf y; y.x = (__bf16)x.x; y.y = (__bf16)x.y;
                y.z = (__bf16)x.z; y.w = (__bf16)x.w;
        *(v4bf*)(out + (size_t)i * 4) = y;
    }
}

// ---------------------------------------------------------------------------
// Kernel 1: Y = act(X @ W^T + b), X:(16384,1024) bf16, W:(1024,1024) bf16.
// Block tile 128x128, K-step 32, double-buffered LDS (80B padded rows ->
// conflict-free b128 fragment reads). Async copies overlap next tile with
// the 8 WMMAs of the current tile. Output stored bf16.
// ---------------------------------------------------------------------------
__global__ __launch_bounds__(256)
void proj_gemm_kernel(const __bf16* __restrict__ X, const __bf16* __restrict__ W,
                      const float* __restrict__ bias, __bf16* __restrict__ Y,
                      int relu) {
    __shared__ __bf16 ldsX[2][128 * 40];
    __shared__ __bf16 ldsW[2][128 * 40];
    const int tid    = threadIdx.x;
    const int lane16 = tid & 15;
    const int hi     = (tid >> 4) & 1;
    const int wave   = tid >> 5;
    const int waveM  = wave & 3;
    const int waveN  = wave >> 2;
    const int mBase  = blockIdx.y * 128;
    const int nBase  = blockIdx.x * 128;

    v8f acc[2][4] = {};

    // 512 16B chunks per tile: chunk c -> row=c>>2 (0..127), col=c&3 (16B unit)
    auto stage = [&](int buf, int kt) {
        #pragma unroll
        for (int j = 0; j < 2; j++) {
            int c   = tid + 256 * j;
            int row = c >> 2;
            int col = c & 3;
            const __bf16* gx = X + (size_t)(mBase + row) * EMB + kt * 32 + col * 8;
            const __bf16* gw = W + (size_t)(nBase + row) * EMB + kt * 32 + col * 8;
            cp16_g2l(gx, &ldsX[buf][row * 40 + col * 8]);
            cp16_g2l(gw, &ldsW[buf][row * 40 + col * 8]);
        }
    };

    stage(0, 0);
    for (int kt = 0; kt < 32; kt++) {
        const int buf = kt & 1;
        cp_fence();
        __syncthreads();
        if (kt < 31) stage(buf ^ 1, kt + 1);

        union { v16bf v; v8bf h2[2]; } afr[2], bfr[4];
        #pragma unroll
        for (int mt = 0; mt < 2; mt++) {
            int row = waveM * 32 + mt * 16 + lane16;
            afr[mt].h2[0] = *(const v8bf*)(&ldsX[buf][row * 40 + hi * 8]);
            afr[mt].h2[1] = *(const v8bf*)(&ldsX[buf][row * 40 + 16 + hi * 8]);
        }
        #pragma unroll
        for (int nt = 0; nt < 4; nt++) {
            int row = waveN * 64 + nt * 16 + lane16;
            bfr[nt].h2[0] = *(const v8bf*)(&ldsW[buf][row * 40 + hi * 8]);
            bfr[nt].h2[1] = *(const v8bf*)(&ldsW[buf][row * 40 + 16 + hi * 8]);
        }
        #pragma unroll
        for (int mt = 0; mt < 2; mt++)
            #pragma unroll
            for (int nt = 0; nt < 4; nt++)
                acc[mt][nt] = __builtin_amdgcn_wmma_f32_16x16x32_bf16(
                    false, afr[mt].v, false, bfr[nt].v,
                    (short)0, acc[mt][nt], false, false);
    }

    #pragma unroll
    for (int mt = 0; mt < 2; mt++) {
        #pragma unroll
        for (int nt = 0; nt < 4; nt++) {
            int col = nBase + waveN * 64 + nt * 16 + lane16;
            float bv = bias[col];
            #pragma unroll
            for (int i = 0; i < 8; i++) {
                int row = mBase + waveM * 32 + mt * 16 + hi * 8 + i;
                float v = acc[mt][nt][i] + bv;
                if (relu) v = fmaxf(v, 0.0f);
                Y[(size_t)row * EMB + col] = (__bf16)v;
            }
        }
    }
}

// ---------------------------------------------------------------------------
// Kernel 2: ksum[n][0:64]=sum_l sin_l*k[n,l,:], [64:128]=cos part (fp32 out).
// ---------------------------------------------------------------------------
__global__ __launch_bounds__(256)
void ksum_kernel(const __bf16* __restrict__ Kp, float* __restrict__ ksum) {
    __shared__ float red[2][256];
    const int n = blockIdx.x, b = n >> 4, h = n & 15;
    const int t = threadIdx.x;
    const int d = t & 63;
    const int part = t >> 6;
    float ps = 0.f, pc = 0.f;
    for (int l = part; l < L_SEQ; l += 4) {
        float kv = (float)Kp[((size_t)l * BATCH + b) * EMB + h * HD + d];
        float s, c; sincos_pos(l, &s, &c);
        ps += s * kv; pc += c * kv;
    }
    red[0][t] = ps; red[1][t] = pc;
    __syncthreads();
    if (t < 64) {
        float s = red[0][t] + red[0][t + 64] + red[0][t + 128] + red[0][t + 192];
        float c = red[1][t] + red[1][t + 64] + red[1][t + 128] + red[1][t + 192];
        ksum[n * 128 + t] = s;
        ksum[n * 128 + 64 + t] = c;
    }
}

// ---------------------------------------------------------------------------
// Kernel 3: kvT[n][m][d'] += sum_l w(l)*k[l,d]*v[l,m]  (d'<64 sin, >=64 cos)
// Scale+transpose during LDS staging; WMMA over K=l; f32 atomic accumulate.
// ---------------------------------------------------------------------------
__global__ __launch_bounds__(256)
void kv_kernel(const __bf16* __restrict__ Kp, const __bf16* __restrict__ Vp,
               float* __restrict__ kvT) {
    __shared__ __bf16 ldsA[128 * 40];   // rows d'=0..127, cols k=l (32)
    __shared__ __bf16 ldsB[64 * 40];    // rows m=0..63,  cols k=l (32)
    const int tid    = threadIdx.x;
    const int lane16 = tid & 15;
    const int hi     = (tid >> 4) & 1;
    const int wave   = tid >> 5;
    const int n = blockIdx.y, b = n >> 4, h = n & 15;
    const int l0 = blockIdx.x * 256;

    v8f acc[4] = {};

    for (int ls = 0; ls < 256; ls += 32) {
        // 256 8-elem chunks: lr = c>>3 (0..31), c8 = c&7 (d-offset 8*c8)
        {
            int lr = tid >> 3;
            int c8 = tid & 7;
            int l  = l0 + ls + lr;
            float s, cc; sincos_pos(l, &s, &cc);
            v8bf kq = *(const v8bf*)(Kp + ((size_t)l * BATCH + b) * EMB + h * HD + c8 * 8);
            v8bf vq = *(const v8bf*)(Vp + ((size_t)l * BATCH + b) * EMB + h * HD + c8 * 8);
            #pragma unroll
            for (int e = 0; e < 8; e++) {
                int dd = c8 * 8 + e;
                float kf = (float)kq[e];
                ldsA[dd * 40 + lr]        = (__bf16)(s * kf);
                ldsA[(64 + dd) * 40 + lr] = (__bf16)(cc * kf);
                ldsB[dd * 40 + lr]        = vq[e];
            }
        }
        __syncthreads();

        union { v16bf v; v8bf h2[2]; } af, bf4;
        int arow = wave * 16 + lane16;
        af.h2[0] = *(const v8bf*)(&ldsA[arow * 40 + hi * 8]);
        af.h2[1] = *(const v8bf*)(&ldsA[arow * 40 + 16 + hi * 8]);
        #pragma unroll
        for (int nt = 0; nt < 4; nt++) {
            int brow = nt * 16 + lane16;
            bf4.h2[0] = *(const v8bf*)(&ldsB[brow * 40 + hi * 8]);
            bf4.h2[1] = *(const v8bf*)(&ldsB[brow * 40 + 16 + hi * 8]);
            acc[nt] = __builtin_amdgcn_wmma_f32_16x16x32_bf16(
                false, af.v, false, bf4.v, (short)0, acc[nt], false, false);
        }
        __syncthreads();
    }

    #pragma unroll
    for (int nt = 0; nt < 4; nt++) {
        int m = nt * 16 + lane16;
        #pragma unroll
        for (int i = 0; i < 8; i++) {
            int dp = wave * 16 + hi * 8 + i;
            atomicAdd(&kvT[((size_t)n * 64 + m) * 128 + dp], acc[nt][i]);
        }
    }
}

// ---------------------------------------------------------------------------
// Kernel 4: out = z * (sin*q | cos*q) @ [kv_sin; kv_cos], fused z-normalize.
// M=128 (l), K=128 (d'), N=64 (m). LDS rows 136 halves (272B, conflict-free).
// ---------------------------------------------------------------------------
__global__ __launch_bounds__(256)
void attn_kernel(const __bf16* __restrict__ Qp, const float* __restrict__ kvT,
                 const float* __restrict__ ksum, float* __restrict__ out) {
    __shared__ __bf16 ldsA[128 * 136];
    __shared__ __bf16 ldsB[64 * 136];
    __shared__ float  zs[128];
    const int tid    = threadIdx.x;
    const int lane16 = tid & 15;
    const int hi     = (tid >> 4) & 1;
    const int wave   = tid >> 5;
    const int n = blockIdx.y, b = n >> 4, h = n & 15;
    const int l0 = blockIdx.x * 128;

    // stage scaled q: 1024 8-elem chunks
    #pragma unroll
    for (int j = 0; j < 4; j++) {
        int c  = tid + 256 * j;
        int lr = c >> 3;             // 0..127
        int c8 = c & 7;
        int l  = l0 + lr;
        float s, cc; sincos_pos(l, &s, &cc);
        v8bf qq = *(const v8bf*)(Qp + ((size_t)l * BATCH + b) * EMB + h * HD + c8 * 8);
        #pragma unroll
        for (int e = 0; e < 8; e++) {
            float qv = (float)qq[e];
            ldsA[lr * 136 + c8 * 8 + e]      = (__bf16)(s * qv);
            ldsA[lr * 136 + 64 + c8 * 8 + e] = (__bf16)(cc * qv);
        }
    }
    // stage kv (fp32 from atomics -> bf16)
    #pragma unroll
    for (int j = 0; j < 8; j++) {
        int idx = tid + 256 * j;     // 0..2047 float4 chunks
        int mr  = idx >> 5;          // 0..63
        int c4  = idx & 31;
        v4f kq = *(const v4f*)(kvT + ((size_t)n * 64 + mr) * 128 + c4 * 4);
        #pragma unroll
        for (int e = 0; e < 4; e++)
            ldsB[mr * 136 + c4 * 4 + e] = (__bf16)(((const float*)&kq)[e]);
    }
    __syncthreads();

    if (tid < 128) {
        float sacc = 0.f;
        const float* ks = ksum + n * 128;
        #pragma unroll 8
        for (int dp = 0; dp < 128; dp++)
            sacc += (float)ldsA[tid * 136 + dp] * ks[dp];
        zs[tid] = 1.0f / fmaxf(sacc, EPS_Z);
    }
    __syncthreads();

    v8f acc[4] = {};
    #pragma unroll
    for (int kk = 0; kk < 128; kk += 32) {
        union { v16bf v; v8bf h2[2]; } af, bf4;
        int arow = wave * 16 + lane16;
        af.h2[0] = *(const v8bf*)(&ldsA[arow * 136 + kk + hi * 8]);
        af.h2[1] = *(const v8bf*)(&ldsA[arow * 136 + kk + 16 + hi * 8]);
        #pragma unroll
        for (int nt = 0; nt < 4; nt++) {
            int brow = nt * 16 + lane16;
            bf4.h2[0] = *(const v8bf*)(&ldsB[brow * 136 + kk + hi * 8]);
            bf4.h2[1] = *(const v8bf*)(&ldsB[brow * 136 + kk + 16 + hi * 8]);
            acc[nt] = __builtin_amdgcn_wmma_f32_16x16x32_bf16(
                false, af.v, false, bf4.v, (short)0, acc[nt], false, false);
        }
    }

    #pragma unroll
    for (int nt = 0; nt < 4; nt++) {
        int m = nt * 16 + lane16;
        #pragma unroll
        for (int i = 0; i < 8; i++) {
            int lr = wave * 16 + hi * 8 + i;
            out[((size_t)(l0 + lr) * BATCH + b) * EMB + h * HD + m]
                = acc[nt][i] * zs[lr];
        }
    }
}

// ---------------------------------------------------------------------------
extern "C" void kernel_launch(void* const* d_in, const int* in_sizes, int n_in,
                              void* d_out, int out_size, void* d_ws, size_t ws_size,
                              hipStream_t stream) {
    (void)in_sizes; (void)n_in; (void)out_size; (void)ws_size;
    const float* query = (const float*)d_in[0];
    const float* key_  = (const float*)d_in[1];
    const float* value = (const float*)d_in[2];
    const float* Wq    = (const float*)d_in[3];
    const float* bq    = (const float*)d_in[4];
    const float* Wk    = (const float*)d_in[5];
    const float* bk    = (const float*)d_in[6];
    const float* Wv    = (const float*)d_in[7];
    const float* bv    = (const float*)d_in[8];

    const size_t nXY = (size_t)ROWS * EMB;   // 16.8M elements
    const size_t nW  = (size_t)EMB * EMB;    // 1M elements

    __bf16* Xq  = (__bf16*)d_ws;
    __bf16* Xk  = Xq  + nXY;
    __bf16* Xv  = Xk  + nXY;
    __bf16* Wqb = Xv  + nXY;
    __bf16* Wkb = Wqb + nW;
    __bf16* Wvb = Wkb + nW;
    __bf16* Qp  = Wvb + nW;
    __bf16* Kp  = Qp  + nXY;
    __bf16* Vp  = Kp  + nXY;
    float*  kvT  = (float*)(Vp + nXY);       // 64*64*128
    float*  ksum = kvT + (size_t)NHEAD * 64 * 128;

    (void)hipMemsetAsync(kvT, 0,
                         (size_t)(NHEAD * 64 * 128 + NHEAD * 128) * sizeof(float),
                         stream);

    // fp32 -> bf16 pre-convert (inputs + weights)
    cvt_kernel<<<dim3((unsigned)(nXY / 1024)), dim3(256), 0, stream>>>(query, Xq, (int)(nXY / 4));
    cvt_kernel<<<dim3((unsigned)(nXY / 1024)), dim3(256), 0, stream>>>(key_,  Xk, (int)(nXY / 4));
    cvt_kernel<<<dim3((unsigned)(nXY / 1024)), dim3(256), 0, stream>>>(value, Xv, (int)(nXY / 4));
    cvt_kernel<<<dim3((unsigned)(nW  / 1024)), dim3(256), 0, stream>>>(Wq, Wqb, (int)(nW / 4));
    cvt_kernel<<<dim3((unsigned)(nW  / 1024)), dim3(256), 0, stream>>>(Wk, Wkb, (int)(nW / 4));
    cvt_kernel<<<dim3((unsigned)(nW  / 1024)), dim3(256), 0, stream>>>(Wv, Wvb, (int)(nW / 4));

    dim3 g1(EMB / 128, ROWS / 128), b1(256);
    proj_gemm_kernel<<<g1, b1, 0, stream>>>(Xq, Wqb, bq, Qp, 1);
    proj_gemm_kernel<<<g1, b1, 0, stream>>>(Xk, Wkb, bk, Kp, 1);
    proj_gemm_kernel<<<g1, b1, 0, stream>>>(Xv, Wvb, bv, Vp, 0);

    ksum_kernel<<<dim3(NHEAD), dim3(256), 0, stream>>>(Kp, ksum);
    kv_kernel  <<<dim3(16, NHEAD), dim3(256), 0, stream>>>(Kp, Vp, kvT);
    attn_kernel<<<dim3(32, NHEAD), dim3(256), 0, stream>>>(Qp, kvT, ksum,
                                                           (float*)d_out);
}
